// DIEN_29214367547615
// MI455X (gfx1250) — matrix operand order repaired and checked by reading, not compiled, not run on hardware
//
#include <hip/hip_runtime.h>
#include <hip/hip_bf16.h>

// ---- problem constants --------------------------------------------------
#define BB 1024
#define SS 200
#define DD 256
#define MM (BB * SS)        // 204800 rows
#define NN 768              // 3 gates * 256

typedef __attribute__((ext_vector_type(16))) _Float16 v16h;
typedef __attribute__((ext_vector_type(8)))  _Float16 v8h;
typedef __attribute__((ext_vector_type(4)))  _Float16 v4h;
typedef __attribute__((ext_vector_type(8)))  float    v8f;

union Frag16 { v16h w; v8h h[2]; };

// ---- K0a: transpose + f32->f16 the six weight matrices ------------------
// Wt[n][k] = W_g[k][n]  (n = g*256 + col), same for U. Row-contiguous in K.
__global__ __launch_bounds__(256) void prep_wt_kernel(
    const float* __restrict__ Wu, const float* __restrict__ Wr,
    const float* __restrict__ Wh, const float* __restrict__ Uu,
    const float* __restrict__ Ur, const float* __restrict__ Uh,
    _Float16* __restrict__ Wt, _Float16* __restrict__ Ut) {
  int blk = blockIdx.x;            // 0..1535
  int set = blk / NN;              // 0 = W, 1 = U
  int n   = blk % NN;              // output row
  int g   = n >> 8;                // gate
  int c   = n & 255;               // source column
  int k   = threadIdx.x;           // source row
  const float* src;
  if (set == 0) src = (g == 0) ? Wu : (g == 1) ? Wr : Wh;
  else          src = (g == 0) ? Uu : (g == 1) ? Ur : Uh;
  _Float16* dst = (set == 0) ? Wt : Ut;
  dst[(size_t)n * DD + k] = (_Float16)src[(size_t)k * DD + c];
}

// ---- K0b: convert x (f32 -> f16), 4 elems / thread ----------------------
__global__ __launch_bounds__(256) void conv_x_kernel(
    const float* __restrict__ x, _Float16* __restrict__ x16, long ntot) {
  long i = ((long)blockIdx.x * blockDim.x + threadIdx.x) * 4;
  if (i + 3 < ntot) {
    float4 xv = *(const float4*)(x + i);
    v4h o = { (_Float16)xv.x, (_Float16)xv.y, (_Float16)xv.z, (_Float16)xv.w };
    *(v4h*)(x16 + i) = o;
  }
}

// ---- K1: v[b,:] = Wa @ item[b]  (v[b,d] = sum_e Wa[d,e]*item[b,e]) ------
__global__ __launch_bounds__(256) void wa_item_kernel(
    const float* __restrict__ Wa, const float* __restrict__ item,
    float* __restrict__ v) {
  __shared__ float it[DD];
  int b = blockIdx.x, d = threadIdx.x;
  it[d] = item[(size_t)b * DD + d];
  __syncthreads();
  const float* wr = Wa + (size_t)d * DD;
  float s = 0.f;
  for (int e = 0; e < DD; e += 4) {
    float4 w = *(const float4*)(wr + e);
    s += w.x * it[e] + w.y * it[e + 1] + w.z * it[e + 2] + w.w * it[e + 3];
  }
  v[(size_t)b * DD + d] = s;
}

// ---- K2: masked softmax attention over S --------------------------------
__global__ __launch_bounds__(256) void attn_kernel(
    const float* __restrict__ x, const float* __restrict__ v,
    const unsigned char* __restrict__ mask, float* __restrict__ attn) {
  __shared__ float vs[DD];
  __shared__ float wred[8];
  __shared__ float bcast;
  int b = blockIdx.x, tid = threadIdx.x;
  int wave = tid >> 5, lane = tid & 31;
  vs[tid] = v[(size_t)b * DD + tid];
  __syncthreads();
  float score = -INFINITY;
  if (tid < SS) {
    const float* xr = x + ((size_t)b * SS + tid) * DD;
    float s = 0.f;
    for (int d = 0; d < DD; d += 4) {
      float4 xv = *(const float4*)(xr + d);
      s += xv.x * vs[d] + xv.y * vs[d + 1] + xv.z * vs[d + 2] + xv.w * vs[d + 3];
    }
    score = mask[(size_t)b * SS + tid] ? s : -INFINITY;
  }
  // block max
  float m = score;
  for (int off = 16; off; off >>= 1) m = fmaxf(m, __shfl_xor(m, off, 32));
  if (lane == 0) wred[wave] = m;
  __syncthreads();
  if (tid == 0) {
    float mm = wred[0];
    for (int i = 1; i < 8; ++i) mm = fmaxf(mm, wred[i]);
    bcast = mm;
  }
  __syncthreads();
  float maxv = bcast;
  float e = (tid < SS && score > -INFINITY) ? __expf(score - maxv) : 0.f;
  // block sum
  float su = e;
  for (int off = 16; off; off >>= 1) su += __shfl_xor(su, off, 32);
  if (lane == 0) wred[wave] = su;
  __syncthreads();
  if (tid == 0) {
    float t = 0.f;
    for (int i = 0; i < 8; ++i) t += wred[i];
    bcast = t;
  }
  __syncthreads();
  float sumv = bcast;
  if (tid < SS)
    attn[(size_t)b * SS + tid] =
        (maxv == -INFINITY) ? (1.f / (float)SS) : e / sumv;
}

// ---- K3: X3[m, 0:768] = x_f16[m,:] @ [Wu|Wr|Wh] + bias  (f16 out) -----
// tile: M=64, N=128 per WG; 8 waves = 4(M) x 2(N); weights staged in LDS
// once, A-fragments read straight from L2-resident x16: barrier-free K loop.
__global__ __launch_bounds__(256) void gemm_x3_kernel(
    const _Float16* __restrict__ x16, const _Float16* __restrict__ Wt,
    const float* __restrict__ bu, const float* __restrict__ br,
    const float* __restrict__ bh, _Float16* __restrict__ X3) {
  __shared__ _Float16 Bs[128][264];  // 256 K + pad -> 528B stride (16B mult)
  const int m0   = blockIdx.x * 64;
  const int n0b  = blockIdx.y * 128;
  const int tid  = threadIdx.x;
  const int wave = tid >> 5, lane = tid & 31;
  const int msub = (wave & 3) * 16;
  const int nsub = (wave >> 2) * 64;
  const int lm   = lane & 15, kh = lane >> 4;

  // stage this WG's 128-row weight slice once
  for (int i = tid; i < 128 * 32; i += 256) {
    int row = i >> 5, seg = i & 31;
    *(v8h*)&Bs[row][seg * 8] =
        *(const v8h*)(Wt + (size_t)(n0b + row) * DD + seg * 8);
  }
  __syncthreads();

  v8f acc[4] = {};
  const _Float16* ap = x16 + (size_t)(m0 + msub + lm) * DD;
#pragma unroll
  for (int kb = 0; kb < 8; ++kb) {
    Frag16 a;
    a.h[0] = *(const v8h*)(ap + kb * 32 + kh * 8);
    a.h[1] = *(const v8h*)(ap + kb * 32 + 16 + kh * 8);
    Frag16 bf[4];
#pragma unroll
    for (int nt = 0; nt < 4; ++nt) {
      int rl = nsub + nt * 16 + lm;
      bf[nt].h[0] = *(const v8h*)&Bs[rl][kb * 32 + kh * 8];
      bf[nt].h[1] = *(const v8h*)&Bs[rl][kb * 32 + 16 + kh * 8];
    }
#pragma unroll
    for (int nt = 0; nt < 4; ++nt)
      acc[nt] = __builtin_amdgcn_wmma_f32_16x16x32_f16(
          false, a.w, false, bf[nt].w, (short)0, acc[nt], false, false);
  }
#pragma unroll
  for (int nt = 0; nt < 4; ++nt) {
    int n = n0b + nsub + nt * 16 + lm;
    const float* bptr = (n < 256) ? bu : (n < 512 ? br : bh);
    float bias = bptr[n & 255];
#pragma unroll
    for (int j = 0; j < 8; ++j) {
      int m = m0 + msub + j + 8 * kh;
      X3[(size_t)m * NN + n] = (_Float16)(acc[nt][j] + bias);
    }
  }
}

// ---- K4: AUGRU recurrence; one WG owns 16 batch rows, loops t=0..199 ----
__global__ __launch_bounds__(256) void augru_scan_kernel(
    const _Float16* __restrict__ X3, const _Float16* __restrict__ Ut,
    const float* __restrict__ attn, float* __restrict__ out) {
  __shared__ _Float16 Hh[16][264];  // f16 copy of h (padded, 528B stride)
  __shared__ float    Hf[16][DD];   // f32 master copy of h
  __shared__ float    G[16][NN];    // h @ [Uu|Ur|Uh]
  const int tid  = threadIdx.x;
  const int wave = tid >> 5, lane = tid & 31;
  const int lm   = lane & 15, kh = lane >> 4;
  const int b0   = blockIdx.x * 16;
  const int em   = tid >> 4;          // elementwise row
  const int ed0  = (tid & 15) * 16;   // elementwise col base

  for (int i = tid; i < 16 * 264; i += 256) (&Hh[0][0])[i] = (_Float16)0.f;
  for (int i = tid; i < 16 * DD;  i += 256) (&Hf[0][0])[i] = 0.f;
  __syncthreads();

  const _Float16* ub = Ut + (size_t)(wave * 96 + lm) * DD + kh * 16;

  for (int t = 0; t < SS; ++t) {
    // --- GEMM phase: 48 N-tiles over 8 waves, A-frag hoisted over q ---
    v8f acc[6] = {};
#pragma unroll
    for (int kb = 0; kb < 8; ++kb) {
      Frag16 a;
      a.h[0] = *(const v8h*)&Hh[lm][kb * 32 + kh * 16];
      a.h[1] = *(const v8h*)&Hh[lm][kb * 32 + kh * 16 + 8];
      Frag16 bf[6];
#pragma unroll
      for (int q = 0; q < 6; ++q) {
        const _Float16* bp = ub + (size_t)(q * 16) * DD + kb * 32;
        bf[q].h[0] = *(const v8h*)bp;
        bf[q].h[1] = *(const v8h*)(bp + 8);
      }
#pragma unroll
      for (int q = 0; q < 6; ++q)
        acc[q] = __builtin_amdgcn_wmma_f32_16x16x32_f16(
            false, a.w, false, bf[q].w, (short)0, acc[q], false, false);
    }
#pragma unroll
    for (int q = 0; q < 6; ++q) {
      int nc = (wave * 6 + q) * 16 + lm;
#pragma unroll
      for (int j = 0; j < 8; ++j) G[j + 8 * kh][nc] = acc[q][j];
    }
    __syncthreads();
    // --- elementwise AUGRU update ---
    {
      int b = b0 + em;
      float a_t = attn[(size_t)b * SS + t];
      size_t xbase = ((size_t)b * SS + t) * NN;
      size_t obase = ((size_t)b * SS + t) * DD;
      if (t + 1 < SS) {  // prefetch next timestep's X3 row segments
        const _Float16* nx = X3 + xbase + NN + ed0;
        __builtin_prefetch(nx, 0, 0);
        __builtin_prefetch(nx + 256, 0, 0);
        __builtin_prefetch(nx + 512, 0, 0);
      }
#pragma unroll 4
      for (int i = 0; i < 16; ++i) {
        int d = ed0 + i;
        float xu = (float)X3[xbase + d];
        float xr = (float)X3[xbase + 256 + d];
        float xh = (float)X3[xbase + 512 + d];
        float gu = G[em][d], gr = G[em][256 + d], gh = G[em][512 + d];
        float hp = Hf[em][d];
        float u  = 1.f / (1.f + __expf(-(xu + gu)));
        float r  = 1.f / (1.f + __expf(-(xr + gr)));
        float hh = tanhf(xh + r * gh);
        float uh = a_t * u;
        float hn = (1.f - uh) * hp + uh * hh;
        Hf[em][d] = hn;
        Hh[em][d] = (_Float16)hn;
        out[obase + d] = hn;
      }
    }
    __syncthreads();
  }
  // h_last appended after outs
  float* hl = out + (size_t)BB * SS * DD;
  for (int i = 0; i < 16; ++i) {
    int d = ed0 + i;
    hl[(size_t)(b0 + em) * DD + d] = Hf[em][d];
  }
}

// ---- launch -------------------------------------------------------------
extern "C" void kernel_launch(void* const* d_in, const int* in_sizes, int n_in,
                              void* d_out, int out_size, void* d_ws,
                              size_t ws_size, hipStream_t stream) {
  const float* x    = (const float*)d_in[0];
  const float* item = (const float*)d_in[1];
  const unsigned char* mask = (const unsigned char*)d_in[2];
  const float* Wa = (const float*)d_in[3];
  const float* Wu = (const float*)d_in[4];
  const float* Uu = (const float*)d_in[5];
  const float* bu = (const float*)d_in[6];
  const float* Wr = (const float*)d_in[7];
  const float* Ur = (const float*)d_in[8];
  const float* br = (const float*)d_in[9];
  const float* Wh = (const float*)d_in[10];
  const float* Uh = (const float*)d_in[11];
  const float* bh = (const float*)d_in[12];
  float* out = (float*)d_out;

  char* ws = (char*)d_ws;
  size_t off = 0;
  auto take = [&](size_t bytes) {
    size_t o = off;
    off = (off + bytes + 255) & ~(size_t)255;
    return o;
  };
  _Float16* x16  = (_Float16*)(ws + take((size_t)MM * DD * 2));
  _Float16* Wt   = (_Float16*)(ws + take((size_t)NN * DD * 2));
  _Float16* Ut   = (_Float16*)(ws + take((size_t)NN * DD * 2));
  float*    v    = (float*)   (ws + take((size_t)BB * DD * 4));
  float*    attn = (float*)   (ws + take((size_t)BB * SS * 4));
  _Float16* X3   = (_Float16*)(ws + take((size_t)MM * NN * 2));

  prep_wt_kernel<<<2 * NN, 256, 0, stream>>>(Wu, Wr, Wh, Uu, Ur, Uh, Wt, Ut);
  long ntot = (long)MM * DD;
  conv_x_kernel<<<(unsigned)(ntot / 4 / 256), 256, 0, stream>>>(x, x16, ntot);
  wa_item_kernel<<<BB, 256, 0, stream>>>(Wa, item, v);
  attn_kernel<<<BB, 256, 0, stream>>>(x, v, mask, attn);
  gemm_x3_kernel<<<dim3(MM / 64, NN / 128), 256, 0, stream>>>(x16, Wt, bu, br,
                                                              bh, X3);
  augru_scan_kernel<<<BB / 16, 256, 0, stream>>>(X3, Ut, attn, out);
}
